// SimplifiedLLaMAMoE_7017976561988
// MI455X (gfx1250) — compile-verified, matrix operand order checked by reading, not run on hardware
//
#include <hip/hip_runtime.h>
#include <hip/hip_bf16.h>
#include <stdint.h>

// ---------------- CDNA5 WMMA types ----------------
typedef __attribute__((ext_vector_type(16))) __bf16 v16bf;
typedef __attribute__((ext_vector_type(8)))  float  v8f;

#define NEXP   8
#define TOPK   2
#define CDIM   2048
#define IDIM   1408
#define ISDIM  2816
#define NTOK   4096      // B*T

#define BM  128
#define BN  64
#define BK  64
#define LDT 72           // padded LDS row stride (bf16): 144B rows, conflict-free b128, 16B aligned

__device__ __forceinline__ unsigned short f2bf(float f) {
  union { float f; unsigned u; } c; c.f = f;
  unsigned r = c.u + 0x7FFFu + ((c.u >> 16) & 1u);  // round-to-nearest-even
  return (unsigned short)(r >> 16);
}

// CDNA5 async copy: global -> LDS, 16B per lane, tracked by ASYNCcnt.
__device__ __forceinline__ void async_b128(unsigned lds_off, const void* gaddr) {
  asm volatile("global_load_async_to_lds_b128 %0, %1, off"
               :: "v"(lds_off), "v"(gaddr) : "memory");
}
__device__ __forceinline__ void wait_async() {
  asm volatile("s_wait_asynccnt 0x0" ::: "memory");
}
__device__ __forceinline__ unsigned lds_off(const void* p) {
  return (unsigned)(uintptr_t)p;   // generic->LDS: low 32 bits are the LDS byte offset
}

// 16x32 bf16 fragment, CDNA5 16-bit A/B layout (ISA 7.12.2):
// lanes 0-15: K 0..7 (VGPR0-3), K 16..23 (VGPR4-7); lanes 16-31: K 8..15, 24..31.
__device__ __forceinline__ v16bf load_frag(const unsigned short* p, int ld) {
  const int lane = threadIdx.x & 31;
  const unsigned short* q = p + (lane & 15) * ld + ((lane >> 4) << 3);
  union { v16bf v; uint4 u[2]; } f;
  f.u[0] = *(const uint4*)(q);
  f.u[1] = *(const uint4*)(q + 16);
  return f.v;
}

__device__ __forceinline__ v8f wmma_bf16(v16bf a, v16bf b, v8f c) {
  return __builtin_amdgcn_wmma_f32_16x16x32_bf16(false, a, false, b, (short)0, c, false, false);
}

// ---------------- small kernels ----------------

// fp32 -> bf16 (vectorized x4)
__global__ __launch_bounds__(256)
void cvt_bf16_kernel(const float* __restrict__ s, unsigned short* __restrict__ d, int n4) {
  const int i = blockIdx.x * 256 + threadIdx.x;
  if (i >= n4) return;
  const size_t j = (size_t)i * 4;
  float4 v = *(const float4*)(s + j);
  union { unsigned short h[4]; uint2 q; } o;
  o.h[0] = f2bf(v.x); o.h[1] = f2bf(v.y); o.h[2] = f2bf(v.z); o.h[3] = f2bf(v.w);
  *(uint2*)(d + j) = o.q;
}

// one wave per token: logits, fp32 softmax, top-2 (ties -> lower index, as lax.top_k)
__global__ __launch_bounds__(128)
void gate_topk_kernel(const float* __restrict__ x, const float* __restrict__ gw,
                      int* __restrict__ tki, float* __restrict__ tkp,
                      int* __restrict__ counts) {
  const int token = blockIdx.x * 4 + (threadIdx.x >> 5);
  const int lane  = threadIdx.x & 31;
  const float* xr = x + (size_t)token * CDIM;
  float lg[NEXP];
#pragma unroll
  for (int e = 0; e < NEXP; ++e) {
    float s = 0.f;
    for (int c = lane; c < CDIM; c += 32) s += xr[c] * gw[e * CDIM + c];
#pragma unroll
    for (int o = 16; o > 0; o >>= 1) s += __shfl_xor(s, o, 32);
    lg[e] = s;
  }
  if (lane == 0) {
    float mx = lg[0];
#pragma unroll
    for (int e = 1; e < NEXP; ++e) mx = fmaxf(mx, lg[e]);
    float p[NEXP], sum = 0.f;
#pragma unroll
    for (int e = 0; e < NEXP; ++e) { p[e] = __expf(lg[e] - mx); sum += p[e]; }
    const float inv = 1.f / sum;
#pragma unroll
    for (int e = 0; e < NEXP; ++e) p[e] *= inv;
    int i0 = 0; float p0 = p[0];
#pragma unroll
    for (int e = 1; e < NEXP; ++e) if (p[e] > p0) { p0 = p[e]; i0 = e; }
    int i1 = -1; float p1 = -1.f;
#pragma unroll
    for (int e = 0; e < NEXP; ++e) if (e != i0 && p[e] > p1) { p1 = p[e]; i1 = e; }
    tki[token * 2 + 0] = i0; tkp[token * 2 + 0] = p0;
    tki[token * 2 + 1] = i1; tkp[token * 2 + 1] = p1;
    atomicAdd(counts + i0, 1);
    atomicAdd(counts + i1, 1);
  }
}

__global__ void prefix_offsets_kernel(const int* __restrict__ counts,
                                      int* __restrict__ offs, int* __restrict__ cursor) {
  if (threadIdx.x == 0) {
    int s = 0;
    for (int e = 0; e < NEXP; ++e) { offs[e] = s; s += counts[e]; cursor[e] = 0; }
    offs[NEXP] = s;
  }
}

__global__ __launch_bounds__(256)
void build_lists_kernel(const int* __restrict__ tki, const float* __restrict__ tkp,
                        const int* __restrict__ offs, int* __restrict__ cursor,
                        int* __restrict__ rows_tok, float* __restrict__ rows_w) {
  const int n = blockIdx.x * 256 + threadIdx.x;
#pragma unroll
  for (int k = 0; k < TOPK; ++k) {
    const int e = tki[n * 2 + k];
    const int pos = atomicAdd(cursor + e, 1);
    rows_tok[offs[e] + pos] = n;
    rows_w  [offs[e] + pos] = tkp[n * 2 + k];
  }
}

// ---------------- GEMM1: H = silu(A @ W1^T) * (A @ W2^T), bf16 in/out ----------------
__global__ __launch_bounds__(256)
void moe_gemm1(const unsigned short* __restrict__ Xb,
               const unsigned short* __restrict__ W1g, const unsigned short* __restrict__ W2g,
               unsigned short* __restrict__ Hout,
               const int* __restrict__ rows_tok, const int* __restrict__ offs,
               int Kdim, int Ncols, int Mshared, int gathered) {
  __shared__ __align__(16) unsigned short lA [2][BM * LDT];
  __shared__ __align__(16) unsigned short lB1[2][BN * LDT];
  __shared__ __align__(16) unsigned short lB2[2][BN * LDT];

  const int e = blockIdx.z;
  int m_base, m_cnt;
  if (gathered) { m_base = offs[e]; m_cnt = offs[e + 1] - m_base; }
  else          { m_base = 0;       m_cnt = Mshared; }
  const int tm = blockIdx.y * BM;
  if (tm >= m_cnt) return;
  const int tn = blockIdx.x * BN;
  const int* toklist = rows_tok + m_base;
  const unsigned short* W1 = W1g + (size_t)e * Ncols * Kdim;
  const unsigned short* W2 = W2g + (size_t)e * Ncols * Kdim;

  const int tid = threadIdx.x;
  const int wm  = tid >> 6;          // 0..3
  const int wn  = (tid >> 5) & 1;    // 0..1

  // ---- per-thread async-copy mapping (fixed rows per block) ----
  const int  arow = tid >> 1;                // one A row per thread
  const int  acol = (tid & 1) * 32;          // elements (half of BK=64)
  const int  gmA  = tm + arow;
  const bool a_ok = (gmA < m_cnt);
  const unsigned short* aptr = Xb;
  if (a_ok) {
    const int src = gathered ? toklist[gmA] : gmA;
    aptr = Xb + (size_t)src * Kdim + acol;
  }
  unsigned aoff[2] = { lds_off(&lA[0][arow * LDT + acol]),
                       lds_off(&lA[1][arow * LDT + acol]) };
  if (!a_ok) {                               // OOB rows: zero once in both buffers
    const uint4 z = make_uint4(0u, 0u, 0u, 0u);
#pragma unroll
    for (int b = 0; b < 2; ++b)
#pragma unroll
      for (int c = 0; c < 32; c += 8)
        *(uint4*)(&lA[b][arow * LDT + acol + c]) = z;
  }
  const int brow = tid >> 2;
  const int bcol = (tid & 3) * 16;
  const unsigned short* b1ptr = W1 + (size_t)(tn + brow) * Kdim + bcol;
  const unsigned short* b2ptr = W2 + (size_t)(tn + brow) * Kdim + bcol;
  unsigned b1off[2] = { lds_off(&lB1[0][brow * LDT + bcol]),
                        lds_off(&lB1[1][brow * LDT + bcol]) };
  unsigned b2off[2] = { lds_off(&lB2[0][brow * LDT + bcol]),
                        lds_off(&lB2[1][brow * LDT + bcol]) };

  auto issue = [&](int kt, int buf) {
    const int ke = kt * BK;
    if (a_ok) {
#pragma unroll
      for (int c = 0; c < 32; c += 8) async_b128(aoff[buf] + c * 2, aptr + ke + c);
    }
#pragma unroll
    for (int c = 0; c < 16; c += 8) {
      async_b128(b1off[buf] + c * 2, b1ptr + ke + c);
      async_b128(b2off[buf] + c * 2, b2ptr + ke + c);
    }
  };

  const v8f zf = {0.f, 0.f, 0.f, 0.f, 0.f, 0.f, 0.f, 0.f};
  v8f acc1[2][2], acc2[2][2];
#pragma unroll
  for (int i = 0; i < 2; ++i)
#pragma unroll
    for (int j = 0; j < 2; ++j) { acc1[i][j] = zf; acc2[i][j] = zf; }

  const int nk = Kdim / BK;
  issue(0, 0);
  wait_async();
  __syncthreads();

  for (int kt = 0; kt < nk; ++kt) {
    const int cur = kt & 1;
    if (kt + 1 < nk) issue(kt + 1, cur ^ 1);   // prefetch next tile (overlaps compute)

#pragma unroll
    for (int kc = 0; kc < BK; kc += 32) {
      const v16bf a0 = load_frag(&lA[cur][(wm * 32 +  0) * LDT + kc], LDT);
      const v16bf a1 = load_frag(&lA[cur][(wm * 32 + 16) * LDT + kc], LDT);
      const v16bf b0 = load_frag(&lB1[cur][(wn * 32 +  0) * LDT + kc], LDT);
      const v16bf b1 = load_frag(&lB1[cur][(wn * 32 + 16) * LDT + kc], LDT);
      acc1[0][0] = wmma_bf16(a0, b0, acc1[0][0]);
      acc1[0][1] = wmma_bf16(a0, b1, acc1[0][1]);
      acc1[1][0] = wmma_bf16(a1, b0, acc1[1][0]);
      acc1[1][1] = wmma_bf16(a1, b1, acc1[1][1]);
      const v16bf c0 = load_frag(&lB2[cur][(wn * 32 +  0) * LDT + kc], LDT);
      const v16bf c1 = load_frag(&lB2[cur][(wn * 32 + 16) * LDT + kc], LDT);
      acc2[0][0] = wmma_bf16(a0, c0, acc2[0][0]);
      acc2[0][1] = wmma_bf16(a0, c1, acc2[0][1]);
      acc2[1][0] = wmma_bf16(a1, c0, acc2[1][0]);
      acc2[1][1] = wmma_bf16(a1, c1, acc2[1][1]);
    }
    wait_async();
    __syncthreads();
  }

  // epilogue: silu-gate, write bf16 H rows (segment-local contiguous)
  const int lane = tid & 31;
  const int nc   = lane & 15;
  const int mr8  = (lane >> 4) << 3;
#pragma unroll
  for (int i = 0; i < 2; ++i)
#pragma unroll
    for (int j = 0; j < 2; ++j)
#pragma unroll
      for (int r = 0; r < 8; ++r) {
        const int m = tm + wm * 32 + i * 16 + mr8 + r;
        if (m >= m_cnt) continue;
        const int n = tn + wn * 32 + j * 16 + nc;
        const float g = acc1[i][j][r];
        const float h = (g / (1.f + __expf(-g))) * acc2[i][j][r];
        Hout[(size_t)(m_base + m) * Ncols + n] = f2bf(h);
      }
}

// ---------------- GEMM2: Y = H @ W3^T; weighted atomic scatter or plain store ----------------
__global__ __launch_bounds__(256)
void moe_gemm2(const unsigned short* __restrict__ Hbuf,
               const unsigned short* __restrict__ W3g, float* __restrict__ Out,
               const int* __restrict__ rows_tok, const float* __restrict__ rows_w,
               const int* __restrict__ offs,
               int Kdim, int Ncols, int Mshared, int gathered) {
  __shared__ __align__(16) unsigned short lA[2][BM * LDT];
  __shared__ __align__(16) unsigned short lB[2][BN * LDT];

  const int e = blockIdx.z;
  int m_base, m_cnt;
  if (gathered) { m_base = offs[e]; m_cnt = offs[e + 1] - m_base; }
  else          { m_base = 0;       m_cnt = Mshared; }
  const int tm = blockIdx.y * BM;
  if (tm >= m_cnt) return;
  const int tn = blockIdx.x * BN;
  const unsigned short* W3 = W3g + (size_t)e * Ncols * Kdim;

  const int tid = threadIdx.x;
  const int wm  = tid >> 6;
  const int wn  = (tid >> 5) & 1;

  const int  arow = tid >> 1;
  const int  acol = (tid & 1) * 32;
  const int  gmA  = tm + arow;
  const bool a_ok = (gmA < m_cnt);
  const unsigned short* aptr = Hbuf;
  if (a_ok) aptr = Hbuf + (size_t)(m_base + gmA) * Kdim + acol;
  unsigned aoff[2] = { lds_off(&lA[0][arow * LDT + acol]),
                       lds_off(&lA[1][arow * LDT + acol]) };
  if (!a_ok) {
    const uint4 z = make_uint4(0u, 0u, 0u, 0u);
#pragma unroll
    for (int b = 0; b < 2; ++b)
#pragma unroll
      for (int c = 0; c < 32; c += 8)
        *(uint4*)(&lA[b][arow * LDT + acol + c]) = z;
  }
  const int brow = tid >> 2;
  const int bcol = (tid & 3) * 16;
  const unsigned short* bptr = W3 + (size_t)(tn + brow) * Kdim + bcol;
  unsigned boff[2] = { lds_off(&lB[0][brow * LDT + bcol]),
                       lds_off(&lB[1][brow * LDT + bcol]) };

  auto issue = [&](int kt, int buf) {
    const int ke = kt * BK;
    if (a_ok) {
#pragma unroll
      for (int c = 0; c < 32; c += 8) async_b128(aoff[buf] + c * 2, aptr + ke + c);
    }
#pragma unroll
    for (int c = 0; c < 16; c += 8) async_b128(boff[buf] + c * 2, bptr + ke + c);
  };

  const v8f zf = {0.f, 0.f, 0.f, 0.f, 0.f, 0.f, 0.f, 0.f};
  v8f acc[2][2];
#pragma unroll
  for (int i = 0; i < 2; ++i)
#pragma unroll
    for (int j = 0; j < 2; ++j) acc[i][j] = zf;

  const int nk = Kdim / BK;
  issue(0, 0);
  wait_async();
  __syncthreads();

  for (int kt = 0; kt < nk; ++kt) {
    const int cur = kt & 1;
    if (kt + 1 < nk) issue(kt + 1, cur ^ 1);

#pragma unroll
    for (int kc = 0; kc < BK; kc += 32) {
      const v16bf a0 = load_frag(&lA[cur][(wm * 32 +  0) * LDT + kc], LDT);
      const v16bf a1 = load_frag(&lA[cur][(wm * 32 + 16) * LDT + kc], LDT);
      const v16bf b0 = load_frag(&lB[cur][(wn * 32 +  0) * LDT + kc], LDT);
      const v16bf b1 = load_frag(&lB[cur][(wn * 32 + 16) * LDT + kc], LDT);
      acc[0][0] = wmma_bf16(a0, b0, acc[0][0]);
      acc[0][1] = wmma_bf16(a0, b1, acc[0][1]);
      acc[1][0] = wmma_bf16(a1, b0, acc[1][0]);
      acc[1][1] = wmma_bf16(a1, b1, acc[1][1]);
    }
    wait_async();
    __syncthreads();
  }

  const int lane = tid & 31;
  const int nc   = lane & 15;
  const int mr8  = (lane >> 4) << 3;
#pragma unroll
  for (int i = 0; i < 2; ++i)
#pragma unroll
    for (int j = 0; j < 2; ++j)
#pragma unroll
      for (int r = 0; r < 8; ++r) {
        const int m = tm + wm * 32 + i * 16 + mr8 + r;
        if (m >= m_cnt) continue;
        const int n = tn + wn * 32 + j * 16 + nc;
        if (gathered) {
          const int   tok = rows_tok[m_base + m];
          const float w   = rows_w  [m_base + m];
          atomicAdd(Out + (size_t)tok * Ncols + n, w * acc[i][j][r]);
        } else {
          Out[(size_t)m * Ncols + n] = acc[i][j][r];
        }
      }
}

// ---------------- launch ----------------
extern "C" void kernel_launch(void* const* d_in, const int* in_sizes, int n_in,
                              void* d_out, int out_size, void* d_ws, size_t ws_size,
                              hipStream_t stream) {
  const float* x   = (const float*)d_in[0];
  const float* gw  = (const float*)d_in[1];
  const float* w1  = (const float*)d_in[2];
  const float* w2  = (const float*)d_in[3];
  const float* w3  = (const float*)d_in[4];
  const float* sw1 = (const float*)d_in[5];
  const float* sw2 = (const float*)d_in[6];
  const float* sw3 = (const float*)d_in[7];
  float* out = (float*)d_out;

  // ---- workspace layout (bytes) ----
  char* ws = (char*)d_ws;
  unsigned short* xb   = (unsigned short*)(ws);                 // 4096*2048*2       = 16,777,216
  unsigned short* hbuf = (unsigned short*)(ws + 16777216);      // 11,534,336 elems  = 23,068,672
  unsigned short* wbf  = (unsigned short*)(ws + 39845888);      // bf16 weights      = 173,015,040
  unsigned short* w1b  = wbf;                                   // 23,068,672 elems
  unsigned short* w2b  = wbf + 23068672;
  unsigned short* w3b  = wbf + 46137344;
  unsigned short* sw1b = wbf + 69206016;                        // 5,767,168 elems
  unsigned short* sw2b = wbf + 74973184;
  unsigned short* sw3b = wbf + 80740352;
  char* meta = ws + 39845888 + 173015040;
  int*   tki      = (int*)  (meta);
  float* tkp      = (float*)(meta + 32768);
  int*   rows_tok = (int*)  (meta + 65536);
  float* rows_w   = (float*)(meta + 98304);
  int*   counts   = (int*)  (meta + 131072);
  int*   offs     = (int*)  (meta + 131072 + 64);
  int*   cursor   = (int*)  (meta + 131072 + 128);

  hipMemsetAsync(counts, 0, NEXP * sizeof(int), stream);

  // one-time bf16 conversions (bf16 weight set = 173 MB, fits the 192 MB L2)
  cvt_bf16_kernel<<<(NTOK * CDIM) / 1024, 256, 0, stream>>>(x, xb, (NTOK * CDIM) / 4);
  cvt_bf16_kernel<<<23068672 / 1024, 256, 0, stream>>>(w1, w1b, 23068672 / 4);
  cvt_bf16_kernel<<<23068672 / 1024, 256, 0, stream>>>(w2, w2b, 23068672 / 4);
  cvt_bf16_kernel<<<23068672 / 1024, 256, 0, stream>>>(w3, w3b, 23068672 / 4);
  cvt_bf16_kernel<<< 5767168 / 1024, 256, 0, stream>>>(sw1, sw1b, 5767168 / 4);
  cvt_bf16_kernel<<< 5767168 / 1024, 256, 0, stream>>>(sw2, sw2b, 5767168 / 4);
  cvt_bf16_kernel<<< 5767168 / 1024, 256, 0, stream>>>(sw3, sw3b, 5767168 / 4);

  gate_topk_kernel<<<NTOK / 4, 128, 0, stream>>>(x, gw, tki, tkp, counts);
  prefix_offsets_kernel<<<1, 32, 0, stream>>>(counts, offs, cursor);
  build_lists_kernel<<<NTOK / 256, 256, 0, stream>>>(tki, tkp, offs, cursor, rows_tok, rows_w);

  // shared expert: writes d_out entirely
  moe_gemm1<<<dim3(ISDIM / BN, NTOK / BM, 1), 256, 0, stream>>>(
      xb, sw1b, sw2b, hbuf, nullptr, nullptr, CDIM, ISDIM, NTOK, 0);
  moe_gemm2<<<dim3(CDIM / BN, NTOK / BM, 1), 256, 0, stream>>>(
      hbuf, sw3b, out, nullptr, nullptr, nullptr, ISDIM, CDIM, NTOK, 0);

  // routed experts: gather-GEMM-scatter (accumulate into d_out)
  moe_gemm1<<<dim3(IDIM / BN, NTOK / BM, NEXP), 256, 0, stream>>>(
      xb, w1b, w2b, hbuf, rows_tok, offs, CDIM, IDIM, 0, 1);
  moe_gemm2<<<dim3(CDIM / BN, NTOK / BM, NEXP), 256, 0, stream>>>(
      hbuf, w3b, out, rows_tok, rows_w, offs, IDIM, CDIM, 0, 1);
}
